// EncoderBlock_85564338471553
// MI455X (gfx1250) — matrix operand order repaired and checked by reading, not compiled, not run on hardware
//
#include <hip/hip_runtime.h>
#include <hip/hip_bf16.h>
#include <math.h>
#include <stdint.h>

// ---------------------------------------------------------------------------
// Problem constants (from reference)
// ---------------------------------------------------------------------------
#define NB 4
#define NS 2048
#define ND 768
#define NH 12
#define NF 3072
#define NDK 64
#define NM (NB * NS) // 8192 rows

typedef _Float16 v8h  __attribute__((ext_vector_type(8)));
typedef _Float16 v16h __attribute__((ext_vector_type(16)));
typedef float    v8f  __attribute__((ext_vector_type(8)));
typedef int      v4i  __attribute__((ext_vector_type(4)));

#define WMMA_F16(a, b, c) \
  __builtin_amdgcn_wmma_f32_16x16x32_f16(false, (a), false, (b), (short)0, (c), false, false)

// CDNA5 async global->LDS path (guarded: falls back to load+ds_store staging)
#if __has_builtin(__builtin_amdgcn_global_load_async_to_lds_b128) && \
    __has_builtin(__builtin_amdgcn_s_wait_asynccnt)
#define USE_ASYNC 1
#define WAIT_ASYNC(n) __builtin_amdgcn_s_wait_asynccnt(n)
typedef __attribute__((address_space(1))) v4i* gas_b128_t;
typedef __attribute__((address_space(3))) v4i* las_b128_t;
#else
#define USE_ASYNC 0
#define WAIT_ASYNC(n)
#endif

__device__ __forceinline__ v16h make_v16(v8h lo, v8h hi) {
  return __builtin_shufflevector(lo, hi, 0, 1, 2, 3, 4, 5, 6, 7, 8, 9, 10, 11, 12, 13, 14, 15);
}

// A fragment: 16x32 f16 (M x K), src at (row0,k0), row-major, leading dim lda.
// Lanes 0-15: row m=lane, K={0..7,16..23}; lanes 16-31: m=lane-16, K={8..15,24..31}.
__device__ __forceinline__ v16h load_frag_a(const _Float16* src, int lda) {
  int l = threadIdx.x & 31;
  int m = l & 15;
  int koff = (l & 16) ? 8 : 0;
  const _Float16* p = src + (size_t)m * lda + koff;
  v8h lo = *(const v8h*)p;
  v8h hi = *(const v8h*)(p + 16);
  return make_v16(lo, hi);
}

// B fragment: 32x16 f16 (K x N) with B[k][n] = W[n][k]; src at (n0,k0) of
// row-major W.  Lanes 0-15: n=lane, K={0..7,16..23}; lanes 16-31: n=lane-16, K={8..15,24..31}.
__device__ __forceinline__ v16h load_frag_b(const _Float16* src, int ldw) {
  int l = threadIdx.x & 31;
  int n = l & 15;
  int koff = (l & 16) ? 8 : 0;
  const _Float16* p = src + (size_t)n * ldw + koff;
  v8h lo = *(const v8h*)p;
  v8h hi = *(const v8h*)(p + 16);
  return make_v16(lo, hi);
}

// ---------------------------------------------------------------------------
// 16-lane max reduction.  DPP16 path: quad_perm xor1/xor2 then row_ror:4/8
// (all-VALU, no LDS round trip); fallback: ds_bpermute-based shfl_xor.
// ---------------------------------------------------------------------------
#if __has_builtin(__builtin_amdgcn_update_dpp)
template <int CTRL>
__device__ __forceinline__ float dpp_maxstep(float v) {
  int s = __builtin_amdgcn_update_dpp(0, __float_as_int(v), CTRL, 0xF, 0xF, true);
  return fmaxf(v, __int_as_float(s));
}
__device__ __forceinline__ float rowmax16(float v) {
  v = dpp_maxstep<0xB1>(v);  // quad_perm [1,0,3,2]  (xor 1)
  v = dpp_maxstep<0x4E>(v);  // quad_perm [2,3,0,1]  (xor 2)
  v = dpp_maxstep<0x124>(v); // row_ror:4
  v = dpp_maxstep<0x128>(v); // row_ror:8
  return v;
}
#else
__device__ __forceinline__ float rowmax16(float v) {
#pragma unroll
  for (int off = 1; off < 16; off <<= 1) v = fmaxf(v, __shfl_xor(v, off, 32));
  return v;
}
#endif

// ---------------------------------------------------------------------------
// f32 -> f16 conversion
// ---------------------------------------------------------------------------
__global__ void f32_to_f16_kernel(const float* __restrict__ in,
                                  _Float16* __restrict__ out, size_t n) {
  size_t i = (size_t)blockIdx.x * blockDim.x + threadIdx.x;
  if (i < n) out[i] = (_Float16)in[i];
}

// ---------------------------------------------------------------------------
// LayerNorm (ddof=1) row kernel -> f16.  block=256, row per block, 3 elem/thr.
// ---------------------------------------------------------------------------
__global__ __launch_bounds__(256) void ln_f16_kernel(
    const float* __restrict__ x, const float* __restrict__ alpha,
    const float* __restrict__ beta, _Float16* __restrict__ y) {
  int row = blockIdx.x;
  int tid = threadIdx.x;
  const float* xr = x + (size_t)row * ND;
  float xv[3];
  float s = 0.f, s2 = 0.f;
#pragma unroll
  for (int j = 0; j < 3; j++) {
    xv[j] = xr[tid + 256 * j];
    s += xv[j];
    s2 += xv[j] * xv[j];
  }
#pragma unroll
  for (int off = 1; off < 32; off <<= 1) {
    s += __shfl_xor(s, off, 32);
    s2 += __shfl_xor(s2, off, 32);
  }
  __shared__ float p0[8], p1[8];
  int w = tid >> 5, l = tid & 31;
  if (l == 0) {
    p0[w] = s;
    p1[w] = s2;
  }
  __syncthreads();
  s = 0.f;
  s2 = 0.f;
#pragma unroll
  for (int i = 0; i < 8; i++) {
    s += p0[i];
    s2 += p1[i];
  }
  float mean = s * (1.f / ND);
  float var = (s2 - (float)ND * mean * mean) * (1.f / (ND - 1));
  float inv = 1.f / (sqrtf(var) + 1e-6f);
  _Float16* yr = y + (size_t)row * ND;
#pragma unroll
  for (int j = 0; j < 3; j++) {
    int i = tid + 256 * j;
    yr[i] = (_Float16)(alpha[i] * (xv[j] - mean) * inv + beta[i]);
  }
}

// ---------------------------------------------------------------------------
// Stage one 64(cols) x 32(k) block of W into LDS (row-major [n][k], ld=32).
// 128 threads x 2 chunks x 16B = 4KB.  Async on CDNA5, fallback otherwise.
// ---------------------------------------------------------------------------
__device__ __forceinline__ void stage_b(const _Float16* W, int col0, int K,
                                        int k0, _Float16* buf, int t) {
#pragma unroll
  for (int c = 0; c < 2; ++c) {
    int chunk = c * 128 + t;
    int row = chunk >> 2;          // 0..63
    int o8 = (chunk & 3) * 8;      // 0,8,16,24 f16 within the 32-k row
    const _Float16* g = W + (size_t)(col0 + row) * K + k0 + o8;
    _Float16* s = buf + row * 32 + o8;
#if USE_ASYNC
    __builtin_amdgcn_global_load_async_to_lds_b128(
        (gas_b128_t)(uintptr_t)g, (las_b128_t)s, 0, 0);
#else
    *(v8h*)s = *(const v8h*)g;
#endif
  }
}

// ---------------------------------------------------------------------------
// WMMA GEMM: out[M,N] = A[M,K] @ W[N,K]^T + bias  (+ optional relu/residual)
// Block = 128 threads (4 waves); wave tile 32x64; block tile 128x64.
// W block double-buffered through LDS via async copies.  grid=(N/64, M/128).
// ---------------------------------------------------------------------------
template <int RELU, int HASRES, int OUT16, int OUT32>
__global__ __launch_bounds__(128) void gemm_wmma_kernel(
    const _Float16* __restrict__ A, const _Float16* __restrict__ W,
    const float* __restrict__ bias, const float* __restrict__ res,
    _Float16* __restrict__ out16, float* __restrict__ out32,
    int M, int N, int K) {
  __shared__ __align__(16) _Float16 bsh[2][64 * 32];
  int t = threadIdx.x;
  int w = t >> 5, l = t & 31;
  int row0 = blockIdx.y * 128 + w * 32;
  int col0 = blockIdx.x * 64;

  v8f acc[2][4];
#pragma unroll
  for (int mi = 0; mi < 2; mi++)
#pragma unroll
    for (int ni = 0; ni < 4; ni++) acc[mi][ni] = {};

  int nk = K >> 5;
  stage_b(W, col0, K, 0, bsh[0], t);
  for (int kt = 0; kt < nk; ++kt) {
    int cur = kt & 1;
    int k0 = kt << 5;
    if (kt + 1 < nk) {
      stage_b(W, col0, K, k0 + 32, bsh[cur ^ 1], t);
      WAIT_ASYNC(2); // current buffer's 2 copies complete (in-order)
    } else {
      WAIT_ASYNC(0);
    }
    __syncthreads();
    v16h a0 = load_frag_a(A + (size_t)row0 * K + k0, K);
    v16h a1 = load_frag_a(A + (size_t)(row0 + 16) * K + k0, K);
#pragma unroll
    for (int ni = 0; ni < 4; ni++) {
      v16h bf = load_frag_b(bsh[cur] + 16 * ni * 32, 32);
      acc[0][ni] = WMMA_F16(a0, bf, acc[0][ni]);
      acc[1][ni] = WMMA_F16(a1, bf, acc[1][ni]);
    }
    __syncthreads(); // protect cur buffer before next-iteration staging
  }

  int rowhalf = (l & 16) ? 8 : 0;
  int lc = l & 15;
  float bcol[4];
#pragma unroll
  for (int ni = 0; ni < 4; ni++) bcol[ni] = bias[col0 + 16 * ni + lc];
#pragma unroll
  for (int mi = 0; mi < 2; mi++) {
#pragma unroll
    for (int ni = 0; ni < 4; ni++) {
#pragma unroll
      for (int r = 0; r < 8; r++) {
        int row = row0 + 16 * mi + r + rowhalf;
        int col = col0 + 16 * ni + lc;
        float v = acc[mi][ni][r] + bcol[ni];
        if (RELU) v = fmaxf(v, 0.f);
        size_t idx = (size_t)row * N + col;
        if (HASRES) v += res[idx];
        if (OUT16) out16[idx] = (_Float16)v;
        if (OUT32) out32[idx] = v;
      }
    }
  }
}

// ---------------------------------------------------------------------------
// Flash attention: one wave per 16-query tile per (b,h); online softmax.
// Row sums via extra WMMA against an all-ones B tile; row max via DPP16.
// Block = 128 threads (4 waves, independent tiles). grid = B*H*(S/16)/4.
// ---------------------------------------------------------------------------
__global__ __launch_bounds__(128) void attn_kernel(
    const _Float16* __restrict__ q, const _Float16* __restrict__ kmat,
    const _Float16* __restrict__ vmat, const int* __restrict__ mask,
    _Float16* __restrict__ ctx) {
  __shared__ __align__(16) _Float16 vt_lds[4][NDK * 32]; // V^T: [dim][key]
  __shared__ __align__(16) _Float16 p_lds[4][16 * 32];   // P:   [query][key]

  int wid = threadIdx.x >> 5;
  int l = threadIdx.x & 31;
  int tile = blockIdx.x * 4 + wid;
  int qt = tile & 127; // S/16 = 128 tiles per (b,h)
  int bh = tile >> 7;
  int h = bh % NH;
  int b = bh / NH;
  int q0 = qt * 16;
  size_t base = ((size_t)b * NS) * ND + (size_t)h * NDK;

  // Q fragments with 1/sqrt(DK)=0.125 pre-folded (exact in f16)
  v16h qa0 = load_frag_a(q + base + (size_t)q0 * ND, ND);
  v16h qa1 = load_frag_a(q + base + (size_t)q0 * ND + 32, ND);
#pragma unroll
  for (int i = 0; i < 16; i++) {
    qa0[i] = qa0[i] * (_Float16)0.125f;
    qa1[i] = qa1[i] * (_Float16)0.125f;
  }

  const v16h vones = {(_Float16)1, (_Float16)1, (_Float16)1, (_Float16)1,
                      (_Float16)1, (_Float16)1, (_Float16)1, (_Float16)1,
                      (_Float16)1, (_Float16)1, (_Float16)1, (_Float16)1,
                      (_Float16)1, (_Float16)1, (_Float16)1, (_Float16)1};

  v8f o0 = {}, o1 = {}, o2 = {}, o3 = {};
  v8f o4 = {}; // running row-sum accumulator (P @ ones)
  float mrun[8];
#pragma unroll
  for (int r = 0; r < 8; r++) mrun[r] = -1e30f;

  _Float16* vt = vt_lds[wid];
  _Float16* pl = p_lds[wid];
  int rowhalf = (l & 16) ? 8 : 0;
  int lc = l & 15;

  for (int kb = 0; kb < NS; kb += 32) {
    __syncthreads();
    { // stage V^T: lane l owns key kb+l (contiguous 64-f16 row read)
      const _Float16* vrow = vmat + base + (size_t)(kb + l) * ND;
#pragma unroll
      for (int c = 0; c < 8; c++) {
        v8h chunk = *(const v8h*)(vrow + c * 8);
#pragma unroll
        for (int e = 0; e < 8; e++) vt[(c * 8 + e) * 32 + l] = chunk[e];
      }
    }
    int mk0 = mask[(size_t)b * NS + kb + lc];
    int mk1 = mask[(size_t)b * NS + kb + 16 + lc];
    __syncthreads();

    v8f sc0 = {}, sc1 = {};
    {
      const _Float16* kb0 = kmat + base + (size_t)kb * ND;
      const _Float16* kb1 = kmat + base + (size_t)(kb + 16) * ND;
      v16h b00 = load_frag_b(kb0, ND);
      v16h b01 = load_frag_b(kb0 + 32, ND);
      v16h b10 = load_frag_b(kb1, ND);
      v16h b11 = load_frag_b(kb1 + 32, ND);
      sc0 = WMMA_F16(qa0, b00, sc0);
      sc0 = WMMA_F16(qa1, b01, sc0);
      sc1 = WMMA_F16(qa0, b10, sc1);
      sc1 = WMMA_F16(qa1, b11, sc1);
    }

#pragma unroll
    for (int r = 0; r < 8; r++) {
      float s0 = (mk0 == 0) ? -1e9f : sc0[r];
      float s1 = (mk1 == 0) ? -1e9f : sc1[r];
      float mx = rowmax16(fmaxf(s0, s1));
      float mn = fmaxf(mrun[r], mx);
      float al = __expf(mrun[r] - mn);
      mrun[r] = mn;
      sc0[r] = __expf(s0 - mn);
      sc1[r] = __expf(s1 - mn);
      o0[r] *= al;
      o1[r] *= al;
      o2[r] *= al;
      o3[r] *= al;
      o4[r] *= al;
    }

#pragma unroll
    for (int r = 0; r < 8; r++) {
      pl[(r + rowhalf) * 32 + lc] = (_Float16)sc0[r];
      pl[(r + rowhalf) * 32 + lc + 16] = (_Float16)sc1[r];
    }
    __syncthreads();

    v16h pa = load_frag_a(pl, 32);
    v16h vb0 = load_frag_b(vt + 0 * 16 * 32, 32);
    v16h vb1 = load_frag_b(vt + 1 * 16 * 32, 32);
    v16h vb2 = load_frag_b(vt + 2 * 16 * 32, 32);
    v16h vb3 = load_frag_b(vt + 3 * 16 * 32, 32);
    o0 = WMMA_F16(pa, vb0, o0);
    o1 = WMMA_F16(pa, vb1, o1);
    o2 = WMMA_F16(pa, vb2, o2);
    o3 = WMMA_F16(pa, vb3, o3);
    o4 = WMMA_F16(pa, vones, o4); // row sums, replicated across columns
  }

#pragma unroll
  for (int r = 0; r < 8; r++) {
    float rl = 1.f / o4[r];
    size_t rowbase = base + (size_t)(q0 + r + rowhalf) * ND;
    ctx[rowbase + 0 * 16 + lc] = (_Float16)(o0[r] * rl);
    ctx[rowbase + 1 * 16 + lc] = (_Float16)(o1[r] * rl);
    ctx[rowbase + 2 * 16 + lc] = (_Float16)(o2[r] * rl);
    ctx[rowbase + 3 * 16 + lc] = (_Float16)(o3[r] * rl);
  }
}

// ---------------------------------------------------------------------------
// Launcher
// ---------------------------------------------------------------------------
extern "C" void kernel_launch(void* const* d_in, const int* in_sizes, int n_in,
                              void* d_out, int out_size, void* d_ws, size_t ws_size,
                              hipStream_t stream) {
  const float* x = (const float*)d_in[0];
  const int* src_mask = (const int*)d_in[1];
  const float* wq = (const float*)d_in[2];
  const float* bq = (const float*)d_in[3];
  const float* wk = (const float*)d_in[4];
  const float* bk = (const float*)d_in[5];
  const float* wv = (const float*)d_in[6];
  const float* bv = (const float*)d_in[7];
  const float* wo = (const float*)d_in[8];
  const float* bo = (const float*)d_in[9];
  const float* w1 = (const float*)d_in[10];
  const float* b1 = (const float*)d_in[11];
  const float* w2 = (const float*)d_in[12];
  const float* b2 = (const float*)d_in[13];
  const float* ln1a = (const float*)d_in[14];
  const float* ln1b = (const float*)d_in[15];
  const float* ln2a = (const float*)d_in[16];
  const float* ln2b = (const float*)d_in[17];
  float* out = (float*)d_out;

  const size_t DD = (size_t)ND * ND;
  const size_t FD = (size_t)NF * ND;
  const size_t MD = (size_t)NM * ND;
  _Float16* wq16 = (_Float16*)d_ws;
  _Float16* wk16 = wq16 + DD;
  _Float16* wv16 = wk16 + DD;
  _Float16* wo16 = wv16 + DD;
  _Float16* w116 = wo16 + DD;
  _Float16* w216 = w116 + FD;
  _Float16* y16 = w216 + FD;
  _Float16* q16 = y16 + MD;
  _Float16* k16 = q16 + MD;
  _Float16* v16 = k16 + MD;
  _Float16* h16 = v16 + MD;
  float* x2 = (float*)k16; // aliases [k16,v16]: dead after attention
  _Float16* ctx16 = y16;   // y dead after QKV projections
  _Float16* y216 = q16;    // q dead after attention

  f32_to_f16_kernel<<<(DD + 255) / 256, 256, 0, stream>>>(wq, wq16, DD);
  f32_to_f16_kernel<<<(DD + 255) / 256, 256, 0, stream>>>(wk, wk16, DD);
  f32_to_f16_kernel<<<(DD + 255) / 256, 256, 0, stream>>>(wv, wv16, DD);
  f32_to_f16_kernel<<<(DD + 255) / 256, 256, 0, stream>>>(wo, wo16, DD);
  f32_to_f16_kernel<<<(FD + 255) / 256, 256, 0, stream>>>(w1, w116, FD);
  f32_to_f16_kernel<<<(FD + 255) / 256, 256, 0, stream>>>(w2, w216, FD);

  ln_f16_kernel<<<NM, 256, 0, stream>>>(x, ln1a, ln1b, y16);

  dim3 gD(ND / 64, NM / 128); // (12, 64)
  gemm_wmma_kernel<0, 0, 1, 0><<<gD, 128, 0, stream>>>(y16, wq16, bq, nullptr, q16, nullptr, NM, ND, ND);
  gemm_wmma_kernel<0, 0, 1, 0><<<gD, 128, 0, stream>>>(y16, wk16, bk, nullptr, k16, nullptr, NM, ND, ND);
  gemm_wmma_kernel<0, 0, 1, 0><<<gD, 128, 0, stream>>>(y16, wv16, bv, nullptr, v16, nullptr, NM, ND, ND);

  attn_kernel<<<(NB * NH * (NS / 16)) / 4, 128, 0, stream>>>(q16, k16, v16, src_mask, ctx16);

  gemm_wmma_kernel<0, 1, 0, 1><<<gD, 128, 0, stream>>>(ctx16, wo16, bo, x, nullptr, x2, NM, ND, ND);

  ln_f16_kernel<<<NM, 256, 0, stream>>>(x2, ln2a, ln2b, y216);

  dim3 gF(NF / 64, NM / 128); // (48, 64)
  gemm_wmma_kernel<1, 0, 1, 0><<<gF, 128, 0, stream>>>(y216, w116, b1, nullptr, h16, nullptr, NM, NF, ND);

  gemm_wmma_kernel<0, 1, 0, 1><<<gD, 128, 0, stream>>>(h16, w216, b2, x2, nullptr, out, NM, ND, NF);
}
